// Model_1778116460895
// MI455X (gfx1250) — matrix operand order repaired and checked by reading, not compiled, not run on hardware
//
#include <hip/hip_runtime.h>

// =====================================================================
// CDNA5 (gfx1250) GConvLSTM-style model, round 3:
//   - weights pre-converted to f16 [O x 64] (transposed/padded) once
//   - m = x @ w_convT                    (WMMA f16, no guards)
//   - scatter-mean over edges            (float4 msgs, f32 atomics in L2)
//   - GRUCell fully fused                (48 WMMA/wave, gates in regs)
//   - LSTM step fully fused              (64 WMMA/wave, dual-A)
//   - out = relu(h1) @ lin_w^T + b       (WMMA, padded O=16, col-guarded)
//   GRU/LSTM weight panels staged to LDS via global_load_async_to_lds_b128
//   (ASYNCcnt + s_wait_asynccnt + barrier); B tiles read from LDS (ds ops).
//   Uniform full-tile fast paths keep EXEC untouched around stores.
// =====================================================================

typedef __attribute__((ext_vector_type(16))) _Float16 v16h;
typedef __attribute__((ext_vector_type(8)))  float    v8f;

#define KDIM 64

#define WMMA_F16(a, b, c) \
  __builtin_amdgcn_wmma_f32_16x16x32_f16(false, (a), false, (b), (short)0, (c), false, false)

// ---- A tile (16x32 f16) from row-major f32 [rows x 64] ----------------
// lanes 0-15 row=lane, K=0..15; lanes 16-31 K=16..31 (contiguous, matches B).
__device__ __forceinline__ v16h load_a_f16(const float* __restrict__ A,
                                           int row_base, int k_base,
                                           int lane, int do_relu) {
  const int r = lane & 15, half = lane >> 4;
  const float* p = A + (size_t)(row_base + r) * KDIM + k_base;
  v16h a;
#pragma unroll
  for (int j = 0; j < 8; ++j) {
    float v0 = p[half * 16 + j];
    float v1 = p[half * 16 + 8 + j];
    if (do_relu) { v0 = fmaxf(v0, 0.f); v1 = fmaxf(v1, 0.f); }
    a[j]     = (_Float16)v0;
    a[8 + j] = (_Float16)v1;
  }
  return a;
}

// ---- B tile (32x16 f16) from f16 weight [Opad x 64] (global or LDS) ----
// B[k][n] = W[col][k]; lanes 0-15 cover K=0..15, lanes 16-31 K=16..31.
__device__ __forceinline__ v16h load_b16(const _Float16* __restrict__ W,
                                         int col, int k_base, int half) {
  return *(const v16h*)(W + (size_t)col * KDIM + k_base + half * 16);
}

// ---- async global->LDS staging (gfx1250 ASYNC path) -------------------
__device__ __forceinline__ void async_copy_b128(void* lds_dst, const void* gsrc) {
  unsigned lds = (unsigned)(size_t)lds_dst;     // addrspace(3) offset in low 32b
  asm volatile("global_load_async_to_lds_b128 %0, %1, off"
               :: "v"(lds), "v"(gsrc) : "memory");
}
__device__ __forceinline__ void stage_weights(_Float16* sdst,
                                              const _Float16* __restrict__ gsrc,
                                              int bytes) {
  // bytes is a multiple of blockDim.x*16 at every callsite -> no divergence
  for (int ofs = threadIdx.x * 16; ofs < bytes; ofs += blockDim.x * 16)
    async_copy_b128((char*)sdst + ofs, (const char*)gsrc + ofs);
  asm volatile("s_wait_asynccnt 0" ::: "memory");
  __syncthreads();
}

__device__ __forceinline__ float sigm(float v) { return 1.f / (1.f + expf(-v)); }

// =====================================================================
// Generic C[N,O] = [relu]A[N,64] @ W[Opad,64]^T + bias   (conv & linear)
// COLGUARD=false when O==Opad (conv): no divergent guards at all.
// =====================================================================
template <bool COLGUARD>
__global__ __launch_bounds__(128) void
gemmT_f16(const float* __restrict__ A, const _Float16* __restrict__ W,
          const float* __restrict__ bias, float* __restrict__ C,
          int Nrows, int O, int Opad, int relu_a) {
  const int lane = threadIdx.x & 31;
  const int wave = threadIdx.x >> 5;
  const int row_base = (blockIdx.x * 4 + wave) * 16;
  if (row_base >= Nrows) return;                 // wave-uniform
  const int n = lane & 15, half = lane >> 4;
  const bool full = (row_base + 16 <= Nrows);    // wave-uniform

  const v16h a0 = load_a_f16(A, row_base, 0, lane, relu_a);
  const v16h a1 = load_a_f16(A, row_base, 32, lane, relu_a);

  for (int c0 = 0; c0 < Opad; c0 += 16) {
    v8f acc = {};
    acc = WMMA_F16(a0, load_b16(W, c0 + n, 0, half), acc);
    acc = WMMA_F16(a1, load_b16(W, c0 + n, 32, half), acc);
    const int col = c0 + n;
    if (!COLGUARD || col < O) {
      const float bv = bias ? bias[col] : 0.f;
      float* cp = C + (size_t)(row_base + half * 8) * O + col;
      if (full) {
#pragma unroll
        for (int i = 0; i < 8; ++i) cp[(size_t)i * O] = acc[i] + bv;
      } else {
#pragma unroll
        for (int i = 0; i < 8; ++i)
          if (row_base + half * 8 + i < Nrows) cp[(size_t)i * O] = acc[i] + bv;
      }
    }
  }
}

// =====================================================================
// Fused GRUCell: h~ = GRU(agg, x).  Weights [wih|whh] (2x192x64 f16)
// staged to LDS (48 KB); 48 WMMAs/wave; gate math in registers.
// =====================================================================
__global__ __launch_bounds__(128) void
gru_fused(const float* __restrict__ agg, const float* __restrict__ x,
          const _Float16* __restrict__ wg,          // [wih | whh] contiguous
          const float* __restrict__ bih, const float* __restrict__ bhh,
          float* __restrict__ ht, int Nrows) {
  __shared__ alignas(32) _Float16 sW[2 * 192 * 64];
  stage_weights(sW, wg, 2 * 192 * 64 * 2);
  const _Float16* sWih = sW;
  const _Float16* sWhh = sW + 192 * 64;

  const int lane = threadIdx.x & 31;
  const int wave = threadIdx.x >> 5;
  const int row_base = (blockIdx.x * 4 + wave) * 16;
  if (row_base >= Nrows) return;                 // after barrier; uniform
  const int n = lane & 15, half = lane >> 4;
  const bool full = (row_base + 16 <= Nrows);

  const v16h aA0 = load_a_f16(agg, row_base, 0, lane, 0);
  const v16h aA1 = load_a_f16(agg, row_base, 32, lane, 0);
  const v16h aX0 = load_a_f16(x, row_base, 0, lane, 0);
  const v16h aX1 = load_a_f16(x, row_base, 32, lane, 0);

#pragma unroll
  for (int t = 0; t < 4; ++t) {
    const int f = t * 16 + n;                    // feature of this lane-column
    v8f gr = {}, gz = {}, gn = {}, hr = {}, hz = {}, hn = {};
    gr = WMMA_F16(aA0, load_b16(sWih, f, 0, half), gr);
    gr = WMMA_F16(aA1, load_b16(sWih, f, 32, half), gr);
    gz = WMMA_F16(aA0, load_b16(sWih, 64 + f, 0, half), gz);
    gz = WMMA_F16(aA1, load_b16(sWih, 64 + f, 32, half), gz);
    gn = WMMA_F16(aA0, load_b16(sWih, 128 + f, 0, half), gn);
    gn = WMMA_F16(aA1, load_b16(sWih, 128 + f, 32, half), gn);
    hr = WMMA_F16(aX0, load_b16(sWhh, f, 0, half), hr);
    hr = WMMA_F16(aX1, load_b16(sWhh, f, 32, half), hr);
    hz = WMMA_F16(aX0, load_b16(sWhh, 64 + f, 0, half), hz);
    hz = WMMA_F16(aX1, load_b16(sWhh, 64 + f, 32, half), hz);
    hn = WMMA_F16(aX0, load_b16(sWhh, 128 + f, 0, half), hn);
    hn = WMMA_F16(aX1, load_b16(sWhh, 128 + f, 32, half), hn);

    const float bi_r = bih[f], bi_z = bih[64 + f], bi_n = bih[128 + f];
    const float bh_r = bhh[f], bh_z = bhh[64 + f], bh_n = bhh[128 + f];
    const size_t base = (size_t)(row_base + half * 8) * 64 + f;
    if (full) {
#pragma unroll
      for (int i = 0; i < 8; ++i) {
        const float rr = sigm(gr[i] + bi_r + hr[i] + bh_r);
        const float zz = sigm(gz[i] + bi_z + hz[i] + bh_z);
        const float nv = tanhf(gn[i] + bi_n + rr * (hn[i] + bh_n));
        ht[base + (size_t)i * 64] = (1.f - zz) * nv + zz * x[base + (size_t)i * 64];
      }
    } else {
#pragma unroll
      for (int i = 0; i < 8; ++i) {
        if (row_base + half * 8 + i >= Nrows) break;
        const float rr = sigm(gr[i] + bi_r + hr[i] + bh_r);
        const float zz = sigm(gz[i] + bi_z + hz[i] + bh_z);
        const float nv = tanhf(gn[i] + bi_n + rr * (hn[i] + bh_n));
        ht[base + (size_t)i * 64] = (1.f - zz) * nv + zz * x[base + (size_t)i * 64];
      }
    }
  }
}

// =====================================================================
// Fused LSTM step: gates = h~@Wih^T + hp@Whh^T (+biases) -> h1, c1.
// Weights [wih|whh] (2x256x64 f16) staged to LDS (64 KB); 64 WMMAs/wave.
// =====================================================================
__global__ __launch_bounds__(128) void
lstm_fused(const float* __restrict__ ht, const float* __restrict__ hp,
           const _Float16* __restrict__ wg,         // [wih | whh] contiguous
           const float* __restrict__ bih, const float* __restrict__ bhh,
           const float* __restrict__ cprev,
           float* __restrict__ h1, float* __restrict__ c1, int Nrows) {
  __shared__ alignas(32) _Float16 sW[2 * 256 * 64];
  stage_weights(sW, wg, 2 * 256 * 64 * 2);
  const _Float16* sWih = sW;
  const _Float16* sWhh = sW + 256 * 64;

  const int lane = threadIdx.x & 31;
  const int wave = threadIdx.x >> 5;
  const int row_base = (blockIdx.x * 4 + wave) * 16;
  if (row_base >= Nrows) return;
  const int n = lane & 15, half = lane >> 4;
  const bool full = (row_base + 16 <= Nrows);

  const v16h aH0 = load_a_f16(ht, row_base, 0, lane, 0);
  const v16h aH1 = load_a_f16(ht, row_base, 32, lane, 0);
  const v16h aP0 = load_a_f16(hp, row_base, 0, lane, 0);
  const v16h aP1 = load_a_f16(hp, row_base, 32, lane, 0);

#pragma unroll
  for (int t = 0; t < 4; ++t) {
    const int f = t * 16 + n;
    v8f gi = {}, gf = {}, gg = {}, go = {};
    gi = WMMA_F16(aH0, load_b16(sWih, f, 0, half), gi);
    gi = WMMA_F16(aH1, load_b16(sWih, f, 32, half), gi);
    gi = WMMA_F16(aP0, load_b16(sWhh, f, 0, half), gi);
    gi = WMMA_F16(aP1, load_b16(sWhh, f, 32, half), gi);
    gf = WMMA_F16(aH0, load_b16(sWih, 64 + f, 0, half), gf);
    gf = WMMA_F16(aH1, load_b16(sWih, 64 + f, 32, half), gf);
    gf = WMMA_F16(aP0, load_b16(sWhh, 64 + f, 0, half), gf);
    gf = WMMA_F16(aP1, load_b16(sWhh, 64 + f, 32, half), gf);
    gg = WMMA_F16(aH0, load_b16(sWih, 128 + f, 0, half), gg);
    gg = WMMA_F16(aH1, load_b16(sWih, 128 + f, 32, half), gg);
    gg = WMMA_F16(aP0, load_b16(sWhh, 128 + f, 0, half), gg);
    gg = WMMA_F16(aP1, load_b16(sWhh, 128 + f, 32, half), gg);
    go = WMMA_F16(aH0, load_b16(sWih, 192 + f, 0, half), go);
    go = WMMA_F16(aH1, load_b16(sWih, 192 + f, 32, half), go);
    go = WMMA_F16(aP0, load_b16(sWhh, 192 + f, 0, half), go);
    go = WMMA_F16(aP1, load_b16(sWhh, 192 + f, 32, half), go);

    const float bi = bih[f] + bhh[f];
    const float bf = bih[64 + f] + bhh[64 + f];
    const float bg = bih[128 + f] + bhh[128 + f];
    const float bo = bih[192 + f] + bhh[192 + f];
    const size_t base = (size_t)(row_base + half * 8) * 64 + f;
    if (full) {
#pragma unroll
      for (int i = 0; i < 8; ++i) {
        const float cv = sigm(gf[i] + bf) * cprev[base + (size_t)i * 64] +
                         sigm(gi[i] + bi) * tanhf(gg[i] + bg);
        h1[base + (size_t)i * 64] = sigm(go[i] + bo) * tanhf(cv);
        c1[base + (size_t)i * 64] = cv;
      }
    } else {
#pragma unroll
      for (int i = 0; i < 8; ++i) {
        if (row_base + half * 8 + i >= Nrows) break;
        const float cv = sigm(gf[i] + bf) * cprev[base + (size_t)i * 64] +
                         sigm(gi[i] + bi) * tanhf(gg[i] + bg);
        h1[base + (size_t)i * 64] = sigm(go[i] + bo) * tanhf(cv);
        c1[base + (size_t)i * 64] = cv;
      }
    }
  }
}

// =====================================================================
// Edge scatter-mean: one thread per (edge, float4 of features).
// b128 message loads; 4 f32 atomics per thread (L2-resident).
// =====================================================================
__global__ void scatter_edges(const float4* __restrict__ m4,
                              const float* __restrict__ ew,
                              const int* __restrict__ src,
                              const int* __restrict__ dst,
                              float* __restrict__ agg, float* __restrict__ cnt,
                              long long total) {       // total = E*16
  long long gid = (long long)blockIdx.x * blockDim.x + threadIdx.x;
  if (gid >= total) return;
  const int e = (int)(gid >> 4);
  const int q = (int)(gid & 15);
  const int s = src[e], d = dst[e];
  const float w = ew[e];
  const float4 v = m4[(size_t)s * 16 + q];
  float* ap = agg + (size_t)d * 64 + q * 4;
  atomicAdd(ap + 0, v.x * w);
  atomicAdd(ap + 1, v.y * w);
  atomicAdd(ap + 2, v.z * w);
  atomicAdd(ap + 3, v.w * w);
  if (q == 0) atomicAdd(&cnt[d], 1.0f);
}

__global__ void zero_f32(float* __restrict__ p, long long n) {
  long long t = (long long)blockIdx.x * blockDim.x + threadIdx.x;
  if (t < n) p[t] = 0.f;
}

// agg (viewed as float4) /= max(cnt, 1)
__global__ void normalize_mean(float4* __restrict__ agg4,
                               const float* __restrict__ cnt, long long n4) {
  long long t = (long long)blockIdx.x * blockDim.x + threadIdx.x;
  if (t >= n4) return;
  const float inv = 1.f / fmaxf(cnt[t >> 4], 1.0f);
  float4 v = agg4[t];
  v.x *= inv; v.y *= inv; v.z *= inv; v.w *= inv;
  agg4[t] = v;
}

// =====================================================================
// Weight pre-conversion (fp32 -> f16, optional zero-pad / transpose)
// =====================================================================
__global__ void cvt_f16_pad(const float* __restrict__ s, _Float16* __restrict__ d,
                            int n_src, int n_tot) {
  int t = blockIdx.x * blockDim.x + threadIdx.x;
  if (t < n_tot) d[t] = (t < n_src) ? (_Float16)s[t] : (_Float16)0.f;
}
__global__ void cvt_f16_transpose64(const float* __restrict__ s,
                                    _Float16* __restrict__ d) {
  int t = blockIdx.x * blockDim.x + threadIdx.x;
  if (t < 64 * 64) { int o = t >> 6, k = t & 63; d[t] = (_Float16)s[k * 64 + o]; }
}

// =====================================================================
extern "C" void kernel_launch(void* const* d_in, const int* in_sizes, int n_in,
                              void* d_out, int out_size, void* d_ws, size_t ws_size,
                              hipStream_t stream) {
  const float* x         = (const float*)d_in[0];
  const float* ew        = (const float*)d_in[1];
  const float* w_conv    = (const float*)d_in[2];
  const float* gru_w_ih  = (const float*)d_in[3];
  const float* gru_w_hh  = (const float*)d_in[4];
  const float* gru_b_ih  = (const float*)d_in[5];
  const float* gru_b_hh  = (const float*)d_in[6];
  const float* lstm_w_ih = (const float*)d_in[7];
  const float* lstm_w_hh = (const float*)d_in[8];
  const float* lstm_b_ih = (const float*)d_in[9];
  const float* lstm_b_hh = (const float*)d_in[10];
  const float* lin_w     = (const float*)d_in[11];
  const float* lin_b     = (const float*)d_in[12];
  const float* h0        = (const float*)d_in[13];
  const float* c0        = (const float*)d_in[14];
  const int*   eidx      = (const int*)d_in[15];

  const int N = in_sizes[0] / 64;
  const int E = in_sizes[1];
  const int* src = eidx;
  const int* dst = eidx + E;

  // ---- workspace: m/h~ | agg | cnt | f16 weights ----------------------
  float* ws    = (float*)d_ws;
  float* m_buf = ws;                          // N*64 (reused as h~)
  float* agg   = ws + (size_t)N * 64;         // N*64
  float* cnt   = ws + (size_t)N * 128;        // N
  size_t fbytes = (((size_t)N * 129) * 4 + 31) & ~(size_t)31;
  _Float16* w_convT = (_Float16*)((char*)d_ws + fbytes);  // 64*64
  _Float16* gruW    = w_convT + 64 * 64;                  // 2*192*64 [wih|whh]
  _Float16* lstmW   = gruW + 2 * 192 * 64;                // 2*256*64 [wih|whh]
  _Float16* linW    = lstmW + 2 * 256 * 64;               // 16*64 (padded)

  float* outp = (float*)d_out;                // N*12
  float* h1   = outp + (size_t)N * 12;        // N*64
  float* c1   = h1 + (size_t)N * 64;          // N*64

  const int tiles = (N + 15) / 16;
  const int gblk  = (tiles + 3) / 4;          // 4 waves (16-row tiles)/block

  // 0) weight conversion (tiny)
  cvt_f16_transpose64<<<16, 256, 0, stream>>>(w_conv, w_convT);
  cvt_f16_pad<<<48, 256, 0, stream>>>(gru_w_ih, gruW, 192 * 64, 192 * 64);
  cvt_f16_pad<<<48, 256, 0, stream>>>(gru_w_hh, gruW + 192 * 64, 192 * 64, 192 * 64);
  cvt_f16_pad<<<64, 256, 0, stream>>>(lstm_w_ih, lstmW, 256 * 64, 256 * 64);
  cvt_f16_pad<<<64, 256, 0, stream>>>(lstm_w_hh, lstmW + 256 * 64, 256 * 64, 256 * 64);
  cvt_f16_pad<<<4, 256, 0, stream>>>(lin_w, linW, 12 * 64, 16 * 64);

  // 1) zero agg + cnt (contiguous N*65 floats)
  zero_f32<<<(int)(((long long)N * 65 + 255) / 256), 256, 0, stream>>>(agg, (long long)N * 65);
  // 2) m = x @ w_conv (unguarded cols: O == Opad)
  gemmT_f16<false><<<gblk, 128, 0, stream>>>(x, w_convT, nullptr, m_buf, N, 64, 64, 0);
  // 3) scatter-add messages + degree counts (float4 per thread)
  {
    long long total = (long long)E * 16;
    scatter_edges<<<(int)((total + 255) / 256), 256, 0, stream>>>(
        (const float4*)m_buf, ew, src, dst, agg, cnt, total);
  }
  // 4) mean normalize (float4)
  {
    long long n4 = (long long)N * 16;
    normalize_mean<<<(int)((n4 + 255) / 256), 256, 0, stream>>>((float4*)agg, cnt, n4);
  }
  // 5) fused GRU -> h~ (overwrites m_buf)
  gru_fused<<<gblk, 128, 0, stream>>>(agg, x, gruW, gru_b_ih, gru_b_hh, m_buf, N);
  // 6) fused LSTM -> h1, c1 (directly into d_out)
  lstm_fused<<<gblk, 128, 0, stream>>>(m_buf, h0, lstmW, lstm_b_ih, lstm_b_hh,
                                       c0, h1, c1, N);
  // 7) out = relu(h1) @ lin_w^T + lin_b (O=12, padded B to 16)
  gemmT_f16<true><<<gblk, 128, 0, stream>>>(h1, linW, lin_b, outp, N, 12, 16, 1);
}
